// QuantumNeuralNetwork_40673340293845
// MI455X (gfx1250) — matrix operand order repaired and checked by reading, not compiled, not run on hardware
//
#include <hip/hip_runtime.h>

typedef _Float16 h8   __attribute__((ext_vector_type(8)));
typedef _Float16 v16h __attribute__((ext_vector_type(16)));
typedef float    v8f  __attribute__((ext_vector_type(8)));
typedef unsigned int u32x4 __attribute__((ext_vector_type(4)));
typedef unsigned int u32x8 __attribute__((ext_vector_type(8)));

#define NB 524288L   // batch
#define DIN 32
#define NQ 4
#define NF 6

// ---- weight image: f16 section (22528 halves) + f32 consts (1072 floats) ----
#define WIMG_HALVES 22528          // W1T(4096) + W2T(16384) + W3T(2048)
#define WIMG_FLOATS 1072           // b1,b2,b3,Wm,bm,Wo1,bo1,Wo2,bo2,pad
#define WIMG_BYTES  (WIMG_HALVES * 2 + WIMG_FLOATS * 4)   // 49344, 8B multiple
// float-section offsets (in floats)
#define OFF_B1  0
#define OFF_B2  128
#define OFF_B3  256
#define OFF_WM  264
#define OFF_BM  392
#define OFF_WO1 424
#define OFF_BO1 936
#define OFF_WO2 952
#define OFF_BO2 1064

#define SMEM_BYTES (WIMG_BYTES + 16384 * 2 + 16384 * 2 + 1024 * 4)  // +H1+H2+CF

__device__ __forceinline__ v16h cat8(h8 lo, h8 hi) {
  return __builtin_shufflevector(lo, hi, 0, 1, 2, 3, 4, 5, 6, 7,
                                 8, 9, 10, 11, 12, 13, 14, 15);
}

// ---- prep: build the contiguous weight image in workspace (once, tiny) ----
__global__ __launch_bounds__(256) void qnn_prep(
    const float* __restrict__ W1, const float* __restrict__ W2,
    const float* __restrict__ W3, const float* __restrict__ b1,
    const float* __restrict__ b2, const float* __restrict__ b3,
    const float* __restrict__ Wm, const float* __restrict__ bm,
    const float* __restrict__ Wo1, const float* __restrict__ bo1,
    const float* __restrict__ Wo2, const float* __restrict__ bo2,
    _Float16* __restrict__ wsH)
{
  const int t = blockIdx.x * blockDim.x + threadIdx.x;
  const int s = gridDim.x * blockDim.x;
  for (int i = t; i < 4096; i += s) {           // W1 (32,128) -> [n][k]
    int n = i >> 5, k = i & 31;
    wsH[i] = (_Float16)W1[k * 128 + n];
  }
  for (int i = t; i < 16384; i += s) {          // W2 (128,128) -> [n][k]
    int n = i >> 7, k = i & 127;
    wsH[4096 + i] = (_Float16)W2[k * 128 + n];
  }
  for (int i = t; i < 2048; i += s) {           // W3 (128,8) -> [16][128], pad n
    int n = i >> 7, k = i & 127;
    wsH[20480 + i] = (n < 8) ? (_Float16)W3[k * 8 + n] : (_Float16)0.0f;
  }
  float* wf = (float*)(wsH + WIMG_HALVES);
  for (int i = t; i < 128; i += s) {
    wf[OFF_B1 + i] = b1[i]; wf[OFF_B2 + i] = b2[i]; wf[OFF_WM + i] = Wm[i];
  }
  for (int i = t; i < 8;   i += s) wf[OFF_B3 + i]  = b3[i];
  for (int i = t; i < 32;  i += s) wf[OFF_BM + i]  = bm[i];
  for (int i = t; i < 512; i += s) wf[OFF_WO1 + i] = Wo1[i];
  for (int i = t; i < 16;  i += s) wf[OFF_BO1 + i] = bo1[i];
  for (int i = t; i < 112; i += s) wf[OFF_WO2 + i] = Wo2[i];
  for (int i = t; i < 7;   i += s) wf[OFF_BO2 + i] = bo2[i];
  if (t == 0) wf[WIMG_FLOATS - 1] = 0.f;
}

template <bool USE_TDM>
__global__ __launch_bounds__(256) void qnn_fused(
    const float* __restrict__ x,
    const float* __restrict__ W1, const float* __restrict__ b1,
    const float* __restrict__ W2, const float* __restrict__ b2,
    const float* __restrict__ W3, const float* __restrict__ b3,
    const float* __restrict__ Wm, const float* __restrict__ bm,
    const float* __restrict__ Wo1, const float* __restrict__ bo1,
    const float* __restrict__ Wo2, const float* __restrict__ bo2,
    const _Float16* __restrict__ wsH, float* __restrict__ out)
{
  extern __shared__ __align__(16) char smem[];
  _Float16* W1T = (_Float16*)smem;           // [128][32]
  _Float16* W2T = W1T + 4096;                // [128][128]
  _Float16* W3T = W2T + 16384;               // [16][128]
  float* CONSTS = (float*)(W3T + 2048);      // 1072 floats (offset 45056)
  _Float16* H1  = (_Float16*)(CONSTS + WIMG_FLOATS);  // offset 49344
  _Float16* H2  = H1 + 16384;
  float* CF     = (float*)(H2 + 16384);      // [128][8]

  float* B1s  = CONSTS + OFF_B1;
  float* B2s  = CONSTS + OFF_B2;
  float* B3s  = CONSTS + OFF_B3;
  float* WmS  = CONSTS + OFF_WM;
  float* bmS  = CONSTS + OFF_BM;
  float* Wo1S = CONSTS + OFF_WO1;
  float* bo1S = CONSTS + OFF_BO1;
  float* Wo2S = CONSTS + OFF_WO2;

  const int tid  = threadIdx.x;
  const int wave = tid >> 5;
  const int lane = tid & 31;
  const int m16  = lane & 15;
  const int hi   = lane >> 4;
  const long rowBase = (long)blockIdx.x * 128;

  if (USE_TDM) {
    // One TDM DMA stages the whole 49344B weight image into LDS (wave 0 only;
    // TDM ignores EXEC, so gate with a wave-uniform branch).
    if (wave == 0) {
      typedef __attribute__((address_space(3))) char lds_char;
      unsigned ldsBase = (unsigned)(unsigned long long)(lds_char*)smem;
      unsigned long long ga = (unsigned long long)wsH;
      const unsigned NE = WIMG_BYTES / 8;      // 6168 8-byte elements
      u32x4 g0;
      g0[0] = 1u;                              // count=1, user descriptor
      g0[1] = ldsBase;                         // lds_addr
      g0[2] = (unsigned)ga;                    // global_addr[31:0]
      g0[3] = (unsigned)((ga >> 32) & 0x01FFFFFFu) | (2u << 30);  // addr|type=2
      u32x8 g1;
      g1[0] = 3u << 16;                        // data_size = 8B
      g1[1] = (NE & 0xFFFFu) << 16;            // tensor_dim0[15:0]
      g1[2] = (NE >> 16) | (1u << 16);         // tensor_dim0[31:16], tensor_dim1=1
      g1[3] = (NE << 16);                      // tile_dim0 = NE
      g1[4] = 0;                               // tile_dim1/2 unused
      g1[5] = NE;                              // tensor_dim0_stride
      g1[6] = 0;
      g1[7] = 0;
      asm volatile("tensor_load_to_lds %0, %1" :: "s"(g0), "s"(g1) : "memory");
      __builtin_amdgcn_s_wait_tensorcnt(0);
    }
  } else {
    // Fallback: elementwise staging straight from the fp32 weights.
    for (int i = tid; i < 4096; i += 256) {
      int n = i >> 5, k = i & 31;
      W1T[i] = (_Float16)W1[k * 128 + n];
    }
    for (int i = tid; i < 16384; i += 256) {
      int n = i >> 7, k = i & 127;
      W2T[i] = (_Float16)W2[k * 128 + n];
    }
    for (int i = tid; i < 2048; i += 256) {
      int n = i >> 7, k = i & 127;
      W3T[i] = (n < 8) ? (_Float16)W3[k * 8 + n] : (_Float16)0.0f;
    }
    if (tid < 128) { B1s[tid] = b1[tid]; B2s[tid] = b2[tid]; WmS[tid] = Wm[tid]; }
    if (tid < 8)   B3s[tid] = b3[tid];
    if (tid < 32)  bmS[tid] = bm[tid];
    for (int i = tid; i < 512; i += 256) Wo1S[i] = Wo1[i];
    if (tid < 16)  bo1S[tid] = bo1[tid];
    if (tid < 112) Wo2S[tid] = Wo2[tid];
    if (tid < 7)   CONSTS[OFF_BO2 + tid] = bo2[tid];
  }
  __syncthreads();

  const int mrow = wave * 16 + m16;

  // ---- GEMM1: h1 = relu(x @ W1 + b1), K = 32; tiles paired to hide the
  //      WMMA->VALU RAW hazard (4 co-exec slots for f16 WMMA) ----
  v16h afrag;
  {
    float abuf[16] __attribute__((aligned(16)));
    const float* xrow = x + (rowBase + mrow) * DIN;
    *(float4*)(abuf + 0)  = ((const float4*)(xrow + hi * 8))[0];
    *(float4*)(abuf + 4)  = ((const float4*)(xrow + hi * 8))[1];
    *(float4*)(abuf + 8)  = ((const float4*)(xrow + 16 + hi * 8))[0];
    *(float4*)(abuf + 12) = ((const float4*)(xrow + 16 + hi * 8))[1];
#pragma unroll
    for (int i = 0; i < 16; ++i) afrag[i] = (_Float16)abuf[i];
  }
#pragma unroll
  for (int c = 0; c < 8; c += 2) {
    const _Float16* bp0 = W1T + ((c + 0) * 16 + m16) * 32 + hi * 8;
    const _Float16* bp1 = W1T + ((c + 1) * 16 + m16) * 32 + hi * 8;
    v16h bf0 = cat8(*(const h8*)bp0, *(const h8*)(bp0 + 16));
    v16h bf1 = cat8(*(const h8*)bp1, *(const h8*)(bp1 + 16));
    v8f acc0 = {0.f, 0.f, 0.f, 0.f, 0.f, 0.f, 0.f, 0.f};
    v8f acc1 = {0.f, 0.f, 0.f, 0.f, 0.f, 0.f, 0.f, 0.f};
    acc0 = __builtin_amdgcn_wmma_f32_16x16x32_f16(false, afrag, false, bf0,
                                                  (short)0, acc0, false, false);
    acc1 = __builtin_amdgcn_wmma_f32_16x16x32_f16(false, afrag, false, bf1,
                                                  (short)0, acc1, false, false);
    float bias0 = B1s[(c + 0) * 16 + m16];
    float bias1 = B1s[(c + 1) * 16 + m16];
    _Float16* hp0 = H1 + (wave * 16 + hi * 8) * 128 + (c + 0) * 16 + m16;
    _Float16* hp1 = H1 + (wave * 16 + hi * 8) * 128 + (c + 1) * 16 + m16;
#pragma unroll
    for (int r = 0; r < 8; ++r) {
      float v = acc0[r] + bias0;
      v = v > 0.f ? v : 0.f;
      hp0[r * 128] = (_Float16)v;
    }
#pragma unroll
    for (int r = 0; r < 8; ++r) {
      float v = acc1[r] + bias1;
      v = v > 0.f ? v : 0.f;
      hp1[r * 128] = (_Float16)v;
    }
  }

  // ---- GEMM2: h2 = relu(h1 @ W2 + b2), K = 128; A fragments hoisted,
  //      two interleaved K-chains hide accumulate + epilogue hazards ----
  v16h a2[4];
#pragma unroll
  for (int kk = 0; kk < 4; ++kk) {
    const _Float16* ap = H1 + mrow * 128 + kk * 32 + hi * 8;
    a2[kk] = cat8(*(const h8*)ap, *(const h8*)(ap + 16));
  }
#pragma unroll
  for (int c = 0; c < 8; c += 2) {
    v8f acc0 = {0.f, 0.f, 0.f, 0.f, 0.f, 0.f, 0.f, 0.f};
    v8f acc1 = {0.f, 0.f, 0.f, 0.f, 0.f, 0.f, 0.f, 0.f};
#pragma unroll
    for (int kk = 0; kk < 4; ++kk) {
      const _Float16* bp0 = W2T + ((c + 0) * 16 + m16) * 128 + kk * 32 + hi * 8;
      const _Float16* bp1 = bp0 + 16 * 128;
      v16h bf0 = cat8(*(const h8*)bp0, *(const h8*)(bp0 + 16));
      v16h bf1 = cat8(*(const h8*)bp1, *(const h8*)(bp1 + 16));
      acc0 = __builtin_amdgcn_wmma_f32_16x16x32_f16(false, a2[kk], false, bf0,
                                                    (short)0, acc0, false, false);
      acc1 = __builtin_amdgcn_wmma_f32_16x16x32_f16(false, a2[kk], false, bf1,
                                                    (short)0, acc1, false, false);
    }
    float bias0 = B2s[(c + 0) * 16 + m16];
    float bias1 = B2s[(c + 1) * 16 + m16];
    _Float16* hp0 = H2 + (wave * 16 + hi * 8) * 128 + (c + 0) * 16 + m16;
    _Float16* hp1 = H2 + (wave * 16 + hi * 8) * 128 + (c + 1) * 16 + m16;
#pragma unroll
    for (int r = 0; r < 8; ++r) {
      float v = acc0[r] + bias0;
      v = v > 0.f ? v : 0.f;
      hp0[r * 128] = (_Float16)v;
    }
#pragma unroll
    for (int r = 0; r < 8; ++r) {
      float v = acc1[r] + bias1;
      v = v > 0.f ? v : 0.f;
      hp1[r * 128] = (_Float16)v;
    }
  }

  // ---- GEMM3: cf = h2 @ W3 + b3 (N padded 8 -> 16) ----
  {
    v8f acc = {0.f, 0.f, 0.f, 0.f, 0.f, 0.f, 0.f, 0.f};
#pragma unroll
    for (int kk = 0; kk < 4; ++kk) {
      const _Float16* ap = H2 + mrow * 128 + kk * 32 + hi * 8;
      v16h af = cat8(*(const h8*)ap, *(const h8*)(ap + 16));
      const _Float16* bp = W3T + m16 * 128 + kk * 32 + hi * 8;
      v16h bf = cat8(*(const h8*)bp, *(const h8*)(bp + 16));
      acc = __builtin_amdgcn_wmma_f32_16x16x32_f16(false, af, false, bf,
                                                   (short)0, acc, false, false);
    }
    if (m16 < 8) {
      float bias = B3s[m16];
      float* cp = CF + (wave * 16 + hi * 8) * 8 + m16;
#pragma unroll
      for (int r = 0; r < 8; ++r) cp[r * 8] = acc[r] + bias;
    }
  }
  __syncthreads();

  // ---- per-row tail: quantum features + small fp32 MLP ----
  if (tid < 128) {
    const long row = rowBase + tid;
    float cf[8] __attribute__((aligned(16)));
#pragma unroll
    for (int i = 0; i < 8; ++i) cf[i] = CF[tid * 8 + i];

    float* out_fault = out;
    float* out_rul   = out + NB * 6;
    float* out_qf    = out + NB * 7;
    float* out_cf    = out + NB * 11;

    ((float4*)(out_cf + row * 8))[0] = *(const float4*)(cf + 0);
    ((float4*)(out_cf + row * 8))[1] = *(const float4*)(cf + 4);

    float q[4] __attribute__((aligned(16)));
    float p = 1.f;
#pragma unroll
    for (int i = 0; i < NQ; ++i) { p *= -__sinf(cf[i]); q[i] = p; }
    ((float4*)(out_qf + row * 4))[0] = *(const float4*)q;

    float meas[32];
#pragma unroll
    for (int j = 0; j < 32; ++j) {
      float s = bmS[j];
#pragma unroll
      for (int i = 0; i < NQ; ++i) s += q[i] * WmS[i * 32 + j];
      meas[j] = s;
    }
    float t1[16];
#pragma unroll
    for (int k = 0; k < 16; ++k) {
      float s = bo1S[k];
#pragma unroll
      for (int j = 0; j < 32; ++j) s += meas[j] * Wo1S[j * 16 + k];
      t1[k] = s > 0.f ? s : 0.f;
    }
#pragma unroll
    for (int u = 0; u < NF + 1; ++u) {
      float s = CONSTS[OFF_BO2 + u];
#pragma unroll
      for (int k = 0; k < 16; ++k) s += t1[k] * Wo2S[k * 7 + u];
      if (u < NF) out_fault[row * 6 + u] = s;
      else        out_rul[row] = s;
    }
  }
}

extern "C" void kernel_launch(void* const* d_in, const int* in_sizes, int n_in,
                              void* d_out, int out_size, void* d_ws, size_t ws_size,
                              hipStream_t stream) {
  (void)in_sizes; (void)n_in; (void)out_size;
  const float* x   = (const float*)d_in[0];
  const float* W1  = (const float*)d_in[1];
  const float* b1  = (const float*)d_in[2];
  const float* W2  = (const float*)d_in[3];
  const float* b2  = (const float*)d_in[4];
  const float* W3  = (const float*)d_in[5];
  const float* b3  = (const float*)d_in[6];
  const float* Wm  = (const float*)d_in[7];
  const float* bm  = (const float*)d_in[8];
  const float* Wo1 = (const float*)d_in[9];
  const float* bo1 = (const float*)d_in[10];
  const float* Wo2 = (const float*)d_in[11];
  const float* bo2 = (const float*)d_in[12];
  float* o = (float*)d_out;
  _Float16* wsH = (_Float16*)d_ws;

  const bool use_tdm = (ws_size >= (size_t)WIMG_BYTES);
  dim3 grid((unsigned)(NB / 128));
  dim3 block(256);

  if (use_tdm) {
    qnn_prep<<<dim3(64), block, 0, stream>>>(W1, W2, W3, b1, b2, b3, Wm, bm,
                                             Wo1, bo1, Wo2, bo2, wsH);
    hipFuncSetAttribute(reinterpret_cast<const void*>(qnn_fused<true>),
                        hipFuncAttributeMaxDynamicSharedMemorySize, SMEM_BYTES);
    qnn_fused<true><<<grid, block, SMEM_BYTES, stream>>>(
        x, W1, b1, W2, b2, W3, b3, Wm, bm, Wo1, bo1, Wo2, bo2, wsH, o);
  } else {
    hipFuncSetAttribute(reinterpret_cast<const void*>(qnn_fused<false>),
                        hipFuncAttributeMaxDynamicSharedMemorySize, SMEM_BYTES);
    qnn_fused<false><<<grid, block, SMEM_BYTES, stream>>>(
        x, W1, b1, W2, b2, W3, b3, Wm, bm, Wo1, bo1, Wo2, bo2, wsH, o);
  }
}
